// NeuralMJDFinancial_75737453298051
// MI455X (gfx1250) — matrix operand order repaired and checked by reading, not compiled, not run on hardware
//
#include <hip/hip_runtime.h>
#include <hip/hip_bf16.h>
#include <stdint.h>

// ---------------------------------------------------------------------------
// NeuralMJD: encoder MLP on v_wmma_f32_16x16x32_f16 (CDNA5 wave32 WMMA),
// then a distribution-equivalent collapsed Euler-Maruyama Monte Carlo.
// ---------------------------------------------------------------------------

typedef __attribute__((ext_vector_type(16))) _Float16 v16h;
typedef __attribute__((ext_vector_type(8)))  float    v8f;

#define LDIM 96
#define DDIM 64
#define HDIM 24
#define WIDN 256
#define KSAMP 1024
#define NPRM (HDIM * DDIM)          // 1536 per-(h,d) parameters

// workspace layout (float offsets)
#define WS_HA   0                   // 64 x 256 activations (ping)
#define WS_HB   16384               // 64 x 256 activations (pong)
#define WS_RAW  32768               // 64 x 128 (raw head, 120 valid cols)
#define WS_PRM  40960               // 5 x 1536 parameter tables

__device__ __forceinline__ float sigm(float v) { return 1.0f / (1.0f + __expf(-v)); }

// C = A(64xK) @ W^T + bias  with optional ReLU.
// W is (N_true x K) row-major, so B[k][n] = W[n*K + k]: rows of W are
// contiguous K-runs -> B fragments are contiguous loads.
// Tiles 16x16, K-steps of 32, striped across the block's 8 wave32s.
__device__ __forceinline__ void gemm_tiles(const float* __restrict__ A, int K,
                                           const float* __restrict__ W,
                                           const float* __restrict__ bias,
                                           float* __restrict__ out, int ldo,
                                           int Ntiles, int Ntrue, bool relu,
                                           int wave, int lane)
{
    const int half = lane >> 4;      // 0: lanes 0-15, 1: lanes 16-31
    const int q    = lane & 15;
    const int tiles = 4 * Ntiles;    // 4 M-tiles (M = 64)
    for (int t = wave; t < tiles; t += 8) {
        const int mt = t / Ntiles, nt = t % Ntiles;
        const int m0 = mt * 16, n0 = nt * 16;
        const int n  = n0 + q;
        v8f c = {};
        for (int k0 = 0; k0 < K; k0 += 32) {
            v16h a, b;
            // A fragment (ISA 16-bit A 16x32): lane holds row m0+q;
            // j<8 -> K = k0+8*half+j ; j>=8 -> K = k0+8*half+16+(j-8)
            const float* arow = A + (m0 + q) * K + k0 + 8 * half;
#pragma unroll
            for (int j = 0; j < 8; ++j) a[j]     = (_Float16)arow[j];
#pragma unroll
            for (int j = 0; j < 8; ++j) a[8 + j] = (_Float16)arow[16 + j];
            // B fragment: col n, same K order as A (j<8 -> k0+8*half+j ; j>=8 -> k0+8*half+16+(j-8))
            if (n < Ntrue) {
                const float* wrow = W + n * K + k0 + 8 * half;
                __builtin_prefetch(wrow + 32, 0, 0);   // global_prefetch_b8
#pragma unroll
                for (int j = 0; j < 8; ++j) { b[j] = (_Float16)wrow[j]; b[8 + j] = (_Float16)wrow[16 + j]; }
            } else {
#pragma unroll
                for (int j = 0; j < 16; ++j) b[j] = (_Float16)0.0f;
            }
            c = __builtin_amdgcn_wmma_f32_16x16x32_f16(
                    false, a, false, b, (short)0, c, false, false);
        }
        if (n < Ntrue) {
            const float bn = bias[n];
#pragma unroll
            for (int r = 0; r < 8; ++r) {
                const int m = m0 + r + 8 * half;     // C/D row layout
                float v = c[r] + bn;
                if (relu) v = fmaxf(v, 0.0f);
                out[m * ldo + n] = v;
            }
        }
    }
}

__global__ void __launch_bounds__(256)
mjd_encoder_kernel(const float* __restrict__ x,
                   const float* __restrict__ W0, const float* __restrict__ b0,
                   const float* __restrict__ W1, const float* __restrict__ b1,
                   const float* __restrict__ W2, const float* __restrict__ b2,
                   const float* __restrict__ W3, const float* __restrict__ b3,
                   float* __restrict__ ws)
{
    __shared__ float xt[DDIM][LDIM];          // 24 KB LDS: x^T  (D x L)
    const int tid  = threadIdx.x;
    const int wave = tid >> 5, lane = tid & 31;

    for (int i = tid; i < LDIM * DDIM; i += 256) {
        const int l = i >> 6, d = i & 63;     // x is (L=96, D=64) row-major
        xt[d][l] = x[i];
    }
    __syncthreads();

    gemm_tiles(&xt[0][0], LDIM, W0, b0, ws + WS_HA, WIDN, 16, 256, true,  wave, lane);
    __threadfence_block(); __syncthreads();
    gemm_tiles(ws + WS_HA, WIDN, W1, b1, ws + WS_HB, WIDN, 16, 256, true,  wave, lane);
    __threadfence_block(); __syncthreads();
    gemm_tiles(ws + WS_HB, WIDN, W2, b2, ws + WS_HA, WIDN, 16, 256, true,  wave, lane);
    __threadfence_block(); __syncthreads();
    gemm_tiles(ws + WS_HA, WIDN, W3, b3, ws + WS_RAW, 128, 8, 120, false, wave, lane);
    __threadfence_block(); __syncthreads();

    // Epilogue: per-channel param tables + cumsum over horizon H.
    // Collapsed EM constants (M*dt == 1 exactly):
    //   base = prev_mean + (mu - lambda*kmjd - 0.5*sigma^2)
    //   diffusion coeff = sigma;  jump: N~Poisson(lambda), nu*N + gamma*sqrt(N)*Z
    if (tid < DDIM) {
        const int d = tid;
        const float* raw = ws + WS_RAW + d * 128;
        float* prm = ws + WS_PRM;
        float run = x[(LDIM - 1) * DDIM + d];            // s0[d]
        for (int h = 0; h < HDIM; ++h) {
            const float mu  = raw[5 * h + 0];
            const float sg  = sigm(raw[5 * h + 1]);
            const float ll  = raw[5 * h + 2];
            const float nu  = tanhf(raw[5 * h + 3]) * 0.5f;
            const float gm  = sigm(raw[5 * h + 4]);
            const float lam = __expf(fminf(ll, 0.0f));
            const float km  = __expf(nu + 0.5f * gm * gm) - 1.0f;
            const float drift = mu - lam * km - 0.5f * sg * sg;
            const int p = h * DDIM + d;
            prm[0 * NPRM + p] = run + drift;             // prev_mean + M*alpha
            prm[1 * NPRM + p] = sg;
            prm[2 * NPRM + p] = nu;
            prm[3 * NPRM + p] = gm;
            prm[4 * NPRM + p] = lam;
            run += mu;                                   // cumsum(mu) over H
        }
    }
}

// ---------------- Monte Carlo: counter-based RNG, 1 thread / output --------

__device__ __forceinline__ uint32_t sm64(uint64_t& s) {
    uint64_t z = (s += 0x9E3779B97F4A7C15ull);
    z = (z ^ (z >> 30)) * 0xBF58476D1CE4E5B9ull;
    z = (z ^ (z >> 27)) * 0x94D049BB133111EBull;
    return (uint32_t)(z ^ (z >> 31));
}
__device__ __forceinline__ float u01(uint64_t& s) {
    return (float)((sm64(s) >> 8) + 1u) * (1.0f / 16777216.0f);   // (0,1]
}

__global__ void __launch_bounds__(256)
mjd_mc_kernel(const float* __restrict__ prm, const int* __restrict__ seedp,
              float* __restrict__ out, int total)
{
    const int idx = blockIdx.x * blockDim.x + threadIdx.x;  // ((k*H)+h)*D+d
    if (idx >= total) return;
    const int p = idx % NPRM;                               // h*64 + d

    const float base = prm[0 * NPRM + p];
    const float sg   = prm[1 * NPRM + p];
    const float nu   = prm[2 * NPRM + p];
    const float gm   = prm[3 * NPRM + p];
    const float lam  = prm[4 * NPRM + p];

    uint64_t s = (((uint64_t)(uint32_t)seedp[0]) << 32)
               ^ (uint64_t)(uint32_t)idx ^ 0xD1B54A32D192ED03ull;
    (void)sm64(s);                                          // warm-up mix

    // Box-Muller pair: z1 -> diffusion, z2 -> jump magnitude
    const float u1 = u01(s), u2 = u01(s);
    const float r  = __fsqrt_rn(-2.0f * __logf(u1));
    float sn, cs;
    __sincosf(6.28318530717958647f * u2, &sn, &cs);
    const float z1 = r * cs, z2 = r * sn;

    // N ~ Poisson(lam), lam <= 1 (Knuth product method, E[iters] ~ 1+lam)
    const float Lth = __expf(-lam);
    float pacc = 1.0f;
    int n = -1;
    do { ++n; pacc *= u01(s); } while (pacc > Lth && n < 64);
    const float nf = (float)n;

    out[idx] = __fmaf_rn(sg, z1, base)
             + __fmaf_rn(nu, nf, gm * __fsqrt_rn(nf) * z2);
}

// ---------------------------------------------------------------------------

extern "C" void kernel_launch(void* const* d_in, const int* in_sizes, int n_in,
                              void* d_out, int out_size, void* d_ws, size_t ws_size,
                              hipStream_t stream) {
    (void)in_sizes; (void)n_in; (void)out_size; (void)ws_size;
    const float* x  = (const float*)d_in[0];
    const float* W0 = (const float*)d_in[1];
    const float* b0 = (const float*)d_in[2];
    const float* W1 = (const float*)d_in[3];
    const float* b1 = (const float*)d_in[4];
    const float* W2 = (const float*)d_in[5];
    const float* b2 = (const float*)d_in[6];
    const float* W3 = (const float*)d_in[7];
    const float* b3 = (const float*)d_in[8];
    const int*  seed = (const int*)d_in[11];
    float* ws  = (float*)d_ws;
    float* out = (float*)d_out;

    mjd_encoder_kernel<<<1, 256, 0, stream>>>(x, W0, b0, W1, b1, W2, b2, W3, b3, ws);

    const int total = KSAMP * HDIM * DDIM;                  // 1,572,864
    mjd_mc_kernel<<<(total + 255) / 256, 256, 0, stream>>>(ws + WS_PRM, seed, out, total);
}
